// FNOnd_56221121905172
// MI455X (gfx1250) — matrix operand order repaired and checked by reading, not compiled, not run on hardware
//
#include <hip/hip_runtime.h>
#include <hip/hip_bf16.h>

// ---------------------------------------------------------------------------
// CDNA5 / gfx1250 implementation of the FNO+attention diffusion block.
// Dense math (attention QK^T, P*V, projections, 3x3 convs as implicit GEMM)
// runs on v_wmma_f32_16x16x32_f16. Weights are pre-converted to fp16 in
// WMMA-A-friendly layout; the GEMM weight panel is staged into LDS with the
// Tensor Data Mover (tensor_load_to_lds + s_wait_tensorcnt) when available.
// Spectral path uses truncated separable DFTs (only 16x16 modes survive).
// ---------------------------------------------------------------------------

typedef __attribute__((ext_vector_type(16))) _Float16 v16h;
typedef __attribute__((ext_vector_type(8)))  _Float16 v8h;
typedef __attribute__((ext_vector_type(8)))  float    v8f;
typedef __attribute__((ext_vector_type(4)))  unsigned int v4u;
typedef __attribute__((ext_vector_type(8)))  int      v8i;
typedef __attribute__((ext_vector_type(4)))  int      v4i;

#define PI_F 3.14159265358979323846f
#define NPIX 4096
#define CCH  64

#ifndef __has_builtin
#define __has_builtin(x) 0
#endif
#if __has_builtin(__builtin_amdgcn_tensor_load_to_lds) && \
    __has_builtin(__builtin_amdgcn_s_wait_tensorcnt)
#define USE_TDM 1
#else
#define USE_TDM 0
#endif

// ---------------------------------------------------------------------------
// WMMA layout (CDNA5 ISA 7.12.2, wave32):
//   A (16x32 f16): lane m = lane&15 ; elem e -> K = e + 8*(e>=8) + 8*(lane>=16)
//   B (32x16 f16): lane n = lane&15 ; elem e -> K = e + 8*(e>=8) + 8*(lane>=16)
//   C/D (16x16 f32): elem r -> m = r + 8*(lane>=16), n = lane&15
// A-operand runs are contiguous 8-element spans -> v8h (b128) loads.
// ---------------------------------------------------------------------------

// --- weight prep: fp32 -> fp16 in (M,K) A-layout ---------------------------
template <int SRC_MK>
__global__ void prep_w_kernel(const float* __restrict__ W, _Float16* __restrict__ Wt)
{
    int idx = blockIdx.x * blockDim.x + threadIdx.x;   // 64*64
    if (idx >= 64 * 64) return;
    int m = idx >> 6, k = idx & 63;
    Wt[idx] = (_Float16)(SRC_MK ? W[m * 64 + k] : W[k * 64 + m]);
}

// conv weights (O,I,3,3) -> fp16 (O, tap*64 + I)
__global__ void prep_convw_kernel(const float* __restrict__ W, _Float16* __restrict__ Wt)
{
    int idx = blockIdx.x * blockDim.x + threadIdx.x;   // 64*576
    if (idx >= 64 * 576) return;
    int m = idx / 576, k = idx - m * 576;
    int t = k >> 6, i = k & 63;
    Wt[idx] = (_Float16)W[m * 576 + i * 9 + t];
}

// --- GEMM: Y[m][n] = sum_k Wt[m][k] * X[k][n] (+bias)  M=64,N=4096,K=64 ----
template <bool HAS_BIAS>
__global__ void gemm_wmma_kernel(const float* __restrict__ X,
                                 const _Float16* __restrict__ Wt,
                                 const float* __restrict__ bias,
                                 float* __restrict__ Y)
{
    __shared__ _Float16 sW[64 * 64];                   // 8 KB fp16 weight panel
#if USE_TDM
    if ((threadIdx.x >> 5) == 0) {
        unsigned lds_off = (unsigned)(unsigned long long)(const void*)&sW[0];
        unsigned long long ga = (unsigned long long)(const void*)Wt;
        v4u g0 = { 1u,                                  // count=1 (valid D#)
                   lds_off,                             // lds_addr
                   (unsigned)ga,                        // global_addr[31:0]
                   (unsigned)((ga >> 32) & 0x1FFFFFFu) | (2u << 30) }; // addr[56:32] | type=2
        v8i g1 = { (int)(1u << 16),                     // data_size=1 (2 bytes)
                   (int)(4096u << 16),                  // tensor_dim0 = 4096 (bits 79:48)
                   (int)(1u << 16),                     // tensor_dim1 = 1    (bits 111:80)
                   (int)(4096u << 16),                  // tile_dim0 = 4096   (bits 127:112)
                   0,                                   // tile_dim1/2 unused
                   4096,                                // tensor_dim0_stride
                   0, 0 };
        v4i gz = { 0, 0, 0, 0 };
        v8i gz8 = { 0, 0, 0, 0, 0, 0, 0, 0 };
        __builtin_amdgcn_tensor_load_to_lds(g0, g1, gz, gz, gz8, 0);
        __builtin_amdgcn_s_wait_tensorcnt((short)0);
    }
#else
    for (int i = threadIdx.x; i < 64 * 64; i += blockDim.x) sW[i] = Wt[i];
#endif
    __syncthreads();

    const int lane = threadIdx.x & 31;
    const int wave = blockIdx.x * 8 + (threadIdx.x >> 5); // 1024 tiles exactly
    const int tm = wave >> 8;                             // 0..3
    const int tn = wave & 255;                            // 0..255
    const int m0  = tm << 4, n0 = tn << 4;
    const int nlo = lane & 15;
    const int hi8 = (lane >= 16) ? 8 : 0;
    const int kb  = (lane >= 16) ? 16 : 0;

    v8f acc = {};
#pragma unroll
    for (int k0 = 0; k0 < 64; k0 += 32) {
        const _Float16* ap = sW + (m0 + nlo) * 64 + k0 + hi8;
        v8h alo = *(const v8h*)ap;                        // K = k0+hi8 .. +7
        v8h ahi = *(const v8h*)(ap + 16);                 // K = k0+16+hi8 .. +7
        v16h a, b;
#pragma unroll
        for (int e = 0; e < 8; ++e) { a[e] = alo[e]; a[8 + e] = ahi[e]; }
        const float* xp = X + (k0 + hi8) * NPIX + n0 + nlo;
#pragma unroll
        for (int e = 0; e < 16; ++e) b[e] = (_Float16)xp[(e + (e & 8)) * NPIX];
        acc = __builtin_amdgcn_wmma_f32_16x16x32_f16(false, a, false, b,
                                                     (short)0, acc, false, false);
    }
#pragma unroll
    for (int r = 0; r < 8; ++r) {
        int mm = m0 + r + hi8;
        float v = acc[r];
        if (HAS_BIAS) v += bias[mm];
        Y[mm * NPIX + n0 + nlo] = v;
    }
}

// --- 3x3 conv (pad 1) as 9-tap implicit GEMM, fp16 weights (O, t*64+i) -----
// Boundary handling is branchless: clamp the tap coordinate (address always
// valid), load unconditionally, then select 0 for out-of-bounds taps.
__global__ void conv3_wmma_kernel(const float* __restrict__ X,
                                  const _Float16* __restrict__ Wt,
                                  const float* __restrict__ bias,
                                  float* __restrict__ Y)
{
    const int lane = threadIdx.x & 31;
    const int wave = blockIdx.x * 8 + (threadIdx.x >> 5);
    const int tm = wave >> 8;
    const int tn = wave & 255;
    const int m0  = tm << 4, n0 = tn << 4;
    const int nlo = lane & 15;
    const int hi8 = (lane >= 16) ? 8 : 0;
    const int kb  = (lane >= 16) ? 16 : 0;
    const int pix = n0 + nlo;
    const int py = pix >> 6, px = pix & 63;
    const _Float16* wrow = Wt + (m0 + nlo) * 576;

    v8f acc = {};
#pragma unroll
    for (int t = 0; t < 9; ++t) {
        const int dy = t / 3 - 1, dx = t % 3 - 1;         // constants (unrolled)
        const int yy = py + dy, xx = px + dx;
        const bool inb = ((unsigned)yy < 64u) && ((unsigned)xx < 64u);
        const int yc = min(max(yy, 0), 63);               // clamped (always valid)
        const int xc = min(max(xx, 0), 63);
        const float zmask = inb ? 1.0f : 0.0f;
        const float* xp = X + (yc << 6) + xc;
#pragma unroll
        for (int k0c = 0; k0c < 64; k0c += 32) {
            const int kbase = t * 64 + k0c;
            const _Float16* ap = wrow + kbase + hi8;
            v8h alo = *(const v8h*)ap;
            v8h ahi = *(const v8h*)(ap + 16);
            v16h a, b;
#pragma unroll
            for (int e = 0; e < 8; ++e) { a[e] = alo[e]; a[8 + e] = ahi[e]; }
            const float* xcp = xp + (k0c + hi8) * 4096;
#pragma unroll
            for (int e = 0; e < 16; ++e)
                b[e] = (_Float16)(xcp[(e + (e & 8)) * 4096] * zmask);  // branchless zero-pad
            acc = __builtin_amdgcn_wmma_f32_16x16x32_f16(false, a, false, b,
                                                         (short)0, acc, false, false);
        }
    }
#pragma unroll
    for (int r = 0; r < 8; ++r) {
        int mm = m0 + r + hi8;
        Y[mm * NPIX + n0 + nlo] = acc[r] + bias[mm];
    }
}

// --- flash attention: one wave per 16-query tile, 32 keys/step -------------
__global__ void attn_wmma_kernel(const float* __restrict__ Q,
                                 const float* __restrict__ Kf,
                                 const float* __restrict__ V,
                                 float* __restrict__ O)
{
    const int N = NPIX;
    __shared__ float pbuf[8][16 * 32];
    const int lane = threadIdx.x & 31;
    const int wid  = threadIdx.x >> 5;
    const int wave = blockIdx.x * 8 + wid;
    float* P = pbuf[wid];
    const int q0   = wave << 4;
    const int nlo  = lane & 15;
    const int hi   = lane >> 4;
    const int hi8  = hi << 3;
    const int kb16 = hi << 4;
    const float scale = 0.125f;

    v16h qa[2];
#pragma unroll
    for (int c2 = 0; c2 < 2; ++c2)
#pragma unroll
        for (int e = 0; e < 16; ++e)
            qa[c2][e] = (_Float16)Q[(c2 * 32 + e + ((e >= 8) ? 8 : 0) + hi8) * N + q0 + nlo];

    float rm[8], rs[8];
    v8f   oacc[4];
#pragma unroll
    for (int r = 0; r < 8; ++r) { rm[r] = -1e30f; rs[r] = 0.0f; }
#pragma unroll
    for (int g = 0; g < 4; ++g) { v8f z = {}; oacc[g] = z; }

    for (int j0 = 0; j0 < N; j0 += 32) {
        if (j0 + 32 < N) {                               // prefetch next K/V tile
            __builtin_prefetch(Kf + lane * N + j0 + 32, 0, 1);
            __builtin_prefetch(V  + lane * N + j0 + 32, 0, 1);
        }
        v8f sArr[2];
#pragma unroll
        for (int h = 0; h < 2; ++h) { v8f z = {}; sArr[h] = z; }
#pragma unroll
        for (int half = 0; half < 2; ++half) {
#pragma unroll
            for (int c2 = 0; c2 < 2; ++c2) {
                v16h kbv;
#pragma unroll
                for (int e = 0; e < 16; ++e)
                    kbv[e] = (_Float16)Kf[(c2 * 32 + hi8 + e + (e & 8)) * N + j0 + half * 16 + nlo];
                sArr[half] = __builtin_amdgcn_wmma_f32_16x16x32_f16(
                    false, qa[c2], false, kbv, (short)0, sArr[half], false, false);
            }
        }
        // online softmax (rows live in 16-lane halves)
        float mx[8], ps[8];
#pragma unroll
        for (int r = 0; r < 8; ++r) {
            float s0 = sArr[0][r] * scale;
            float s1 = sArr[1][r] * scale;
            sArr[0][r] = s0; sArr[1][r] = s1;
            mx[r] = fmaxf(s0, s1);
        }
#pragma unroll
        for (int mask = 1; mask <= 8; mask <<= 1)
#pragma unroll
            for (int r = 0; r < 8; ++r)
                mx[r] = fmaxf(mx[r], __shfl_xor(mx[r], mask, 32));
#pragma unroll
        for (int r = 0; r < 8; ++r) {
            float mnew = fmaxf(rm[r], mx[r]);
            float corr = __expf(rm[r] - mnew);
            rm[r] = mnew;
            float p0 = __expf(sArr[0][r] - mnew);
            float p1 = __expf(sArr[1][r] - mnew);
            ps[r] = p0 + p1;
            int m = r + hi8;
            P[m * 32 + nlo]      = p0;
            P[m * 32 + 16 + nlo] = p1;
            rs[r] *= corr;
#pragma unroll
            for (int g = 0; g < 4; ++g) oacc[g][r] *= corr;
        }
#pragma unroll
        for (int mask = 1; mask <= 8; mask <<= 1)
#pragma unroll
            for (int r = 0; r < 8; ++r)
                ps[r] += __shfl_xor(ps[r], mask, 32);
#pragma unroll
        for (int r = 0; r < 8; ++r) rs[r] += ps[r];
        __syncthreads();
        // P (C-layout in LDS) -> A operand; contiguous 8-float runs (ds b128)
        const float4* pp0 = (const float4*)(P + nlo * 32 + hi8);
        const float4* pp1 = (const float4*)(P + nlo * 32 + 16 + hi8);
        float4 pq0 = pp0[0], pq1 = pp0[1], pq2 = pp1[0], pq3 = pp1[1];
        float pv[16] = { pq0.x, pq0.y, pq0.z, pq0.w, pq1.x, pq1.y, pq1.z, pq1.w,
                         pq2.x, pq2.y, pq2.z, pq2.w, pq3.x, pq3.y, pq3.z, pq3.w };
        v16h pa;
#pragma unroll
        for (int e = 0; e < 16; ++e) pa[e] = (_Float16)pv[e];
#pragma unroll
        for (int g = 0; g < 4; ++g) {
            const float4* vp = (const float4*)(V + (g * 16 + nlo) * N + j0 + hi8);
            float4 v0 = vp[0], v1 = vp[1], v2 = vp[4], v3 = vp[5];
            float vvv[16] = { v0.x, v0.y, v0.z, v0.w, v1.x, v1.y, v1.z, v1.w,
                              v2.x, v2.y, v2.z, v2.w, v3.x, v3.y, v3.z, v3.w };
            v16h vb;
#pragma unroll
            for (int e = 0; e < 16; ++e) vb[e] = (_Float16)vvv[e];
            oacc[g] = __builtin_amdgcn_wmma_f32_16x16x32_f16(
                false, pa, false, vb, (short)0, oacc[g], false, false);
        }
        __syncthreads();
    }
    float inv[8];
#pragma unroll
    for (int r = 0; r < 8; ++r) inv[r] = 1.0f / rs[r];
#pragma unroll
    for (int g = 0; g < 4; ++g) {
        float4 o0 = { oacc[g][0] * inv[0], oacc[g][1] * inv[1],
                      oacc[g][2] * inv[2], oacc[g][3] * inv[3] };
        float4 o1 = { oacc[g][4] * inv[4], oacc[g][5] * inv[5],
                      oacc[g][6] * inv[6], oacc[g][7] * inv[7] };
        float* op = O + (g * 16 + nlo) * N + q0 + hi8;
        *(float4*)op = o0;
        *(float4*)(op + 4) = o1;
    }
}

// ------------------------- spectral (truncated DFT) ------------------------

__global__ void dftx_kernel(const float* __restrict__ X, float* __restrict__ T)
{
    int idx = blockIdx.x * blockDim.x + threadIdx.x;       // (c,y,l): 64*64*16
    if (idx >= 64 * 64 * 16) return;
    int l = idx & 15, y = (idx >> 4) & 63, c = idx >> 10;
    const float* row = X + c * 4096 + y * 64;
    float re = 0.f, im = 0.f;
    for (int x = 0; x < 64; ++x) {
        float ang = -(PI_F / 32.0f) * (float)(l * x);
        float s, co; __sincosf(ang, &s, &co);
        re += row[x] * co; im += row[x] * s;
    }
    T[2 * idx] = re * 0.125f; T[2 * idx + 1] = im * 0.125f;
}

__global__ void dfty_kernel(const float* __restrict__ T, float* __restrict__ Xf)
{
    int idx = blockIdx.x * blockDim.x + threadIdx.x;       // (c,k,l): 64*16*16
    if (idx >= 64 * 16 * 16) return;
    int l = idx & 15, k = (idx >> 4) & 15, c = idx >> 8;
    float re = 0.f, im = 0.f;
    for (int y = 0; y < 64; ++y) {
        float ang = -(PI_F / 32.0f) * (float)(k * y);
        float s, co; __sincosf(ang, &s, &co);
        int ti = 2 * ((c * 64 + y) * 16 + l);
        float tr = T[ti], tim = T[ti + 1];
        re += tr * co - tim * s;
        im += tr * s + tim * co;
    }
    Xf[2 * idx] = re * 0.125f; Xf[2 * idx + 1] = im * 0.125f;
}

__global__ void modemix_kernel(const float* __restrict__ Xf,
                               const float* __restrict__ wre,
                               const float* __restrict__ wim,
                               float* __restrict__ Of)
{
    int idx = blockIdx.x * blockDim.x + threadIdx.x;       // (o,k,l): 64*16*16
    if (idx >= 64 * 256) return;
    int l = idx & 15, k = (idx >> 4) & 15, o = idx >> 8;
    float re = 0.f, im = 0.f;
    for (int i = 0; i < 64; ++i) {
        int xi = 2 * ((i << 8) + (k << 4) + l);
        float xr = Xf[xi], xim = Xf[xi + 1];
        int wi = ((i * 64 + o) * 16 + k) * 16 + l;
        float wr = wre[wi], wiv = wim[wi];
        re += xr * wr - xim * wiv;
        im += xr * wiv + xim * wr;
    }
    Of[2 * idx] = re; Of[2 * idx + 1] = im;
}

__global__ void idfty_kernel(const float* __restrict__ Of, float* __restrict__ G)
{
    int idx = blockIdx.x * blockDim.x + threadIdx.x;       // (o,y,l): 64*64*16
    if (idx >= 64 * 64 * 16) return;
    int l = idx & 15, y = (idx >> 4) & 63, o = idx >> 10;
    float re = 0.f, im = 0.f;
    for (int k = 0; k < 16; ++k) {
        float ang = (PI_F / 32.0f) * (float)(k * y);
        float s, co; __sincosf(ang, &s, &co);
        int oi = 2 * ((o << 8) + (k << 4) + l);
        float fr = Of[oi], fi = Of[oi + 1];
        re += fr * co - fi * s;
        im += fr * s + fi * co;
    }
    G[2 * idx] = re * 0.125f; G[2 * idx + 1] = im * 0.125f;
}

__global__ void idftx_kernel(const float* __restrict__ G, float* __restrict__ S)
{
    int idx = blockIdx.x * blockDim.x + threadIdx.x;       // (o,y,x): 64*4096
    if (idx >= 64 * 4096) return;
    int x = idx & 63, y = (idx >> 6) & 63, o = idx >> 12;
    const float* g = G + 2 * ((o * 64 + y) * 16);
    float acc = g[0];                                      // l = 0 (real part)
    for (int l = 1; l < 16; ++l) {
        float ang = (PI_F / 32.0f) * (float)(l * x);
        float s, co; __sincosf(ang, &s, &co);
        acc += 2.0f * (g[2 * l] * co - g[2 * l + 1] * s);
    }
    S[idx] = acc * 0.125f;
}

// ------------------------------ small kernels ------------------------------

__global__ void temb_kernel(const int* __restrict__ t,
                            const float* __restrict__ w1, const float* __restrict__ b1,
                            const float* __restrict__ w2, const float* __restrict__ b2,
                            float* __restrict__ out)
{
    __shared__ float e0[64], h[64];
    int i = threadIdx.x;                                   // 64 threads
    float tv = (float)t[0];
    int j = i & 31;
    float fr  = __expf(-logf(10000.0f) * (float)j / 32.0f);
    float arg = tv * fr;
    e0[i] = (i < 32) ? __sinf(arg) : __cosf(arg);
    __syncthreads();
    float a = b1[i];
    for (int k = 0; k < 64; ++k) a += e0[k] * w1[k * 64 + i];
    float g = 0.5f * a * (1.0f + tanhf(0.7978845608f * (a + 0.044715f * a * a * a)));
    h[i] = g;
    __syncthreads();
    float a2 = b2[i];
    for (int k = 0; k < 64; ++k) a2 += h[k] * w2[k * 64 + i];
    out[i] = a2;
}

__global__ void lift_kernel(const float* __restrict__ x, const float* __restrict__ w,
                            const float* __restrict__ b, float* __restrict__ y)
{
    int idx = blockIdx.x * blockDim.x + threadIdx.x;       // 64*4096
    if (idx >= 64 * 4096) return;
    int o = idx >> 12, n = idx & 4095;
    float acc = b[o];
    for (int i = 0; i < 4; ++i) acc += x[i * 4096 + n] * w[i * 64 + o];
    y[idx] = acc;
}

__global__ void film_kernel(const float* __restrict__ temb,
                            const float* __restrict__ sw, const float* __restrict__ sb,
                            const float* __restrict__ hw, const float* __restrict__ hb,
                            float* __restrict__ scaleV, float* __restrict__ shiftV)
{
    int c = threadIdx.x;                                   // 64 threads
    float a = sb[c], d = hb[c];
    for (int j = 0; j < 64; ++j) {
        float tv = temb[j];
        a += tv * sw[j * 64 + c];
        d += tv * hw[j * 64 + c];
    }
    scaleV[c] = a; shiftV[c] = d;
}

__global__ void norm_film_relu_kernel(const float* __restrict__ X,
                                      const float* __restrict__ gamma,
                                      const float* __restrict__ beta,
                                      const float* __restrict__ scaleV,
                                      const float* __restrict__ shiftV,
                                      float* __restrict__ Y)
{
    int n = blockIdx.x * blockDim.x + threadIdx.x;         // 4096 pixels
    if (n >= 4096) return;
    float mu = 0.f;
    for (int c = 0; c < 64; ++c) mu += X[c * 4096 + n];
    mu *= (1.0f / 64.0f);
    float var = 0.f;
    for (int c = 0; c < 64; ++c) { float d = X[c * 4096 + n] - mu; var += d * d; }
    var *= (1.0f / 64.0f);
    float inv = rsqrtf(var + 1e-5f);
    for (int c = 0; c < 64; ++c) {
        float xn = (X[c * 4096 + n] - mu) * inv * gamma[c] + beta[c];
        float v  = xn * scaleV[c] + shiftV[c];
        Y[c * 4096 + n] = fmaxf(v, 0.0f);
    }
}

__global__ void mul_kernel(const float* __restrict__ a, const float* __restrict__ b,
                           float* __restrict__ o)
{
    int i = blockIdx.x * blockDim.x + threadIdx.x;
    if (i < 64 * 4096) o[i] = a[i] * b[i];
}

__global__ void sigmul_kernel(const float* __restrict__ m, const float* __restrict__ f,
                              float* __restrict__ o)
{
    int i = blockIdx.x * blockDim.x + threadIdx.x;
    if (i < 64 * 4096) o[i] = m[i] * (1.0f / (1.0f + __expf(-f[i])));
}

__global__ void proj_kernel(const float* __restrict__ X, const float* __restrict__ pw,
                            const float* __restrict__ pb, float* __restrict__ out)
{
    int n = blockIdx.x * blockDim.x + threadIdx.x;         // 4096
    if (n >= 4096) return;
    float acc = pb[0];
    for (int i = 0; i < 64; ++i) acc += X[i * 4096 + n] * pw[i];
    out[n] = acc;
}

// ------------------------------ host driver --------------------------------

struct AttnP { const float *bo, *wk, *wo, *wq, *wv; };
struct BlkP  { const float *beta, *cb, *cw, *gamma, *hb, *hw, *sb, *sw; };
struct NbpP  { BlkP blk[6]; const float *final_b, *final_w, *init_b, *init_w; };
struct AFP   { AttnP attn; NbpP nbp; const float *w_im, *w_re; };

extern "C" void kernel_launch(void* const* d_in, const int* in_sizes, int n_in,
                              void* d_out, int out_size, void* d_ws, size_t ws_size,
                              hipStream_t stream)
{
    (void)in_sizes; (void)out_size;
    const float* diff = (const float*)d_in[0];
    const float* cond = (const float*)d_in[1];
    const int*   tptr = (const int*)d_in[2];

    // --- parameter pytree walk (JAX: dict keys sorted alphabetically) -----
    int pi = 3;
    const float* big = (n_in <= 8) ? (const float*)d_in[3] : nullptr;
    size_t bigOff = 0;
    auto leaf = [&](size_t cnt) -> const float* {
        if (big) { const float* p = big + bigOff; bigOff += cnt; return p; }
        return (const float*)d_in[pi++];
    };

    AFP af[2][2][2];
    for (int oc = 0; oc < 2; ++oc)
        for (int b = 0; b < 2; ++b)
            for (int f = 0; f < 2; ++f) {
                AFP& a = af[oc][b][f];
                a.attn.bo = leaf(64);   a.attn.wk = leaf(4096);
                a.attn.wo = leaf(4096); a.attn.wq = leaf(4096);
                a.attn.wv = leaf(4096);
                for (int k = 0; k < 6; ++k) {
                    BlkP& bl = a.nbp.blk[k];
                    bl.beta = leaf(64);    bl.cb = leaf(64);
                    bl.cw   = leaf(36864); bl.gamma = leaf(64);
                    bl.hb   = leaf(64);    bl.hw = leaf(4096);
                    bl.sb   = leaf(64);    bl.sw = leaf(4096);
                }
                a.nbp.final_b = leaf(64);    a.nbp.final_w = leaf(4096);
                a.nbp.init_b  = leaf(64);    a.nbp.init_w  = leaf(36864);
                a.w_im = leaf(1048576);      a.w_re = leaf(1048576);
            }
    const float* lift_b = leaf(64);
    const float* lift_w = leaf(256);
    const float* proj_b = leaf(1);
    const float* proj_w = leaf(64);
    const float* tm_b1  = leaf(64);
    const float* tm_b2  = leaf(64);
    const float* tm_w1  = leaf(4096);
    const float* tm_w2  = leaf(4096);

    // --- workspace layout (fp32 elements) ---------------------------------
    const size_t FM = (size_t)CCH * NPIX;                  // 262144
    float* ws = (float*)d_ws;
    const size_t need = 256 + 14 * FM + 131072 + 32768 + 32768 + 131072 + 18432;
    if (ws_size < need * sizeof(float)) return;
    float* tembBuf = ws;          // 64
    float* filmS   = ws + 64;     // 64
    float* filmH   = ws + 128;    // 64
    float* base    = ws + 256;
    float* d0      = base + 0 * FM;
    float* c0      = base + 1 * FM;
    float* bFormer = base + 2 * FM;
    float* bCb     = base + 3 * FM;
    float* bQ      = base + 4 * FM;
    float* bK      = base + 5 * FM;
    float* bV      = base + 6 * FM;
    float* bO      = base + 7 * FM;
    float* bAttn   = base + 8 * FM;
    float* bS1     = base + 9 * FM;
    float* bS2     = base + 10 * FM;
    float* bM      = base + 11 * FM;
    float* bX      = base + 12 * FM;
    float* bY      = base + 13 * FM;
    float* bT      = base + 14 * FM;         // 131072
    float* bXf     = bT + 131072;            // 32768
    float* bOf     = bXf + 32768;            // 32768
    float* bG      = bOf + 32768;            // 131072
    _Float16* bWt  = (_Float16*)(bG + 131072);  // 36864 halves (18432 floats)
    float* outF    = (float*)d_out;

    // --- launch helpers ----------------------------------------------------
    auto gemm = [&](const float* X, const float* Wsrc, const float* bias,
                    float* Y, int srcMK) {
        if (srcMK) prep_w_kernel<1><<<16, 256, 0, stream>>>(Wsrc, bWt);
        else       prep_w_kernel<0><<<16, 256, 0, stream>>>(Wsrc, bWt);
        if (bias)  gemm_wmma_kernel<true ><<<128, 256, 0, stream>>>(X, bWt, bias, Y);
        else       gemm_wmma_kernel<false><<<128, 256, 0, stream>>>(X, bWt, nullptr, Y);
    };
    auto conv3 = [&](const float* X, const float* Wsrc, const float* bias, float* Y) {
        prep_convw_kernel<<<144, 256, 0, stream>>>(Wsrc, bWt);
        conv3_wmma_kernel<<<128, 256, 0, stream>>>(X, bWt, bias, Y);
    };
    auto spec = [&](const float* x, const float* wre, const float* wim, float* S) {
        dftx_kernel   <<<256,  256, 0, stream>>>(x, bT);
        dfty_kernel   <<<64,   256, 0, stream>>>(bT, bXf);
        modemix_kernel<<<64,   256, 0, stream>>>(bXf, wre, wim, bOf);
        idfty_kernel  <<<256,  256, 0, stream>>>(bOf, bG);
        idftx_kernel  <<<1024, 256, 0, stream>>>(bG, S);
    };
    auto attfno = [&](const float* Qc, const float* Kc, const AFP& p, float* out) {
        gemm(Qc, p.attn.wq, nullptr, bQ, 0);
        gemm(Kc, p.attn.wk, nullptr, bK, 0);
        gemm(Kc, p.attn.wv, nullptr, bV, 0);
        attn_wmma_kernel<<<32, 256, 0, stream>>>(bQ, bK, bV, bO);
        gemm(bO, p.attn.wo, p.attn.bo, bAttn, 0);
        spec(bQ, p.w_re, p.w_im, bS1);                     // spec(Qs)
        spec(bK, p.w_re, p.w_im, bS2);                     // spec(Ks)
        mul_kernel<<<1024, 256, 0, stream>>>(bS1, bS2, bM);
        conv3(bM, p.nbp.init_w, p.nbp.init_b, bX);
        for (int k = 0; k < 6; ++k) {
            const BlkP& bl = p.nbp.blk[k];
            conv3(bX, bl.cw, bl.cb, bY);
            film_kernel<<<1, 64, 0, stream>>>(tembBuf, bl.sw, bl.sb, bl.hw, bl.hb,
                                              filmS, filmH);
            norm_film_relu_kernel<<<16, 256, 0, stream>>>(bY, bl.gamma, bl.beta,
                                                          filmS, filmH, bX);
        }
        gemm(bX, p.nbp.final_w, p.nbp.final_b, bY, 1);     // 1x1 conv (O,I) layout
        sigmul_kernel<<<1024, 256, 0, stream>>>(bM, bY, bS1);  // M * sigmoid(filt)
        mul_kernel<<<1024, 256, 0, stream>>>(bS1, bAttn, out); // * attn_out
    };

    // --- forward -----------------------------------------------------------
    temb_kernel<<<1, 64, 0, stream>>>(tptr, tm_w1, tm_b1, tm_w2, tm_b2, tembBuf);
    lift_kernel<<<1024, 256, 0, stream>>>(diff, lift_w, lift_b, d0);
    lift_kernel<<<1024, 256, 0, stream>>>(cond, lift_w, lift_b, c0);

    for (int oc = 0; oc < 2; ++oc) {
        const float* cbPtr = c0;
        for (int b = 0; b < 2; ++b) {
            attfno(cbPtr, d0, af[oc][b][0], bFormer);      // f1: Qc=cond, Kc=diff
            attfno(d0, bFormer, af[oc][b][1], bCb);        // f2: Qc=diff, Kc=former
            cbPtr = bCb;
        }
        proj_kernel<<<16, 256, 0, stream>>>(cbPtr, proj_w, proj_b, outF + oc * NPIX);
    }
}
